// net_72808285602319
// MI455X (gfx1250) — compile-verified, matrix operand order checked by reading, not run on hardware
//
#include <hip/hip_runtime.h>
#include <hip/hip_bf16.h>
#include <stdint.h>

typedef __bf16 bf16_t;
typedef __attribute__((ext_vector_type(8)))  __bf16 bfx8;
typedef __attribute__((ext_vector_type(16))) __bf16 bfx16;
typedef __attribute__((ext_vector_type(8)))  float  v8f;

#define BM 128
#define BN 128
#define BK 32

#define AS1 __attribute__((address_space(1)))
#define AS3 __attribute__((address_space(3)))

// Matches the builtin's parameter type reported by clang:
//   '__attribute__((__vector_size__(2 * sizeof(int)))) int __device__ *'
typedef int async_b64_t __attribute__((vector_size(2 * sizeof(int))));

#if __has_builtin(__builtin_amdgcn_global_load_async_to_lds_b64)
#define USE_ASYNC_LDS 1
#endif

#if __has_builtin(__builtin_amdgcn_s_wait_asynccnt)
#define WAIT_ASYNC() __builtin_amdgcn_s_wait_asynccnt(0)
#else
#define WAIT_ASYNC() asm volatile("s_wait_asynccnt 0" ::: "memory")
#endif

// ---------------------------------------------------------------------------
// fp32 -> bf16 streaming convert
// ---------------------------------------------------------------------------
__global__ void k_cvt_bf16(const float* __restrict__ s, bf16_t* __restrict__ d, int n) {
  int i = blockIdx.x * blockDim.x + threadIdx.x;
  int stride = gridDim.x * blockDim.x;
  for (; i < n; i += stride) d[i] = (bf16_t)s[i];
}

// fp32 W[K][N] -> bf16 Wt[N][K]  (transpose + convert, 32x32 LDS tiles)
__global__ void k_cvtT_bf16(const float* __restrict__ src, bf16_t* __restrict__ dst,
                            int K, int N) {
  __shared__ float tile[32][33];
  int k0 = blockIdx.y * 32, n0 = blockIdx.x * 32;
  int tx = threadIdx.x & 31, ty = threadIdx.x >> 5;   // 32x8 threads
  for (int r = ty; r < 32; r += 8) {
    int k = k0 + r, n = n0 + tx;
    tile[r][tx] = (k < K && n < N) ? src[(size_t)k * N + n] : 0.f;
  }
  __syncthreads();
  for (int r = ty; r < 32; r += 8) {
    int n = n0 + r, k = k0 + tx;
    if (n < N && k < K) dst[(size_t)n * K + k] = (bf16_t)tile[tx][r];
  }
}

__global__ void k_zero_f32(float* __restrict__ p, int n) {
  int i = blockIdx.x * blockDim.x + threadIdx.x;
  int stride = gridDim.x * blockDim.x;
  for (; i < n; i += stride) p[i] = 0.f;
}

// ---------------------------------------------------------------------------
// bf16 WMMA GEMM:  C[M,N] = act(A[M,K] * Bt[N,K]^T + bias)
// A row-major bf16, Bt row-major bf16 (i.e. B pre-transposed).
// Writes fp32 Cf and/or bf16 Cb.
// ---------------------------------------------------------------------------

// Full 32-wide K tile, rows guarded by EXEC only (stale LDS for OOB rows is
// harmless: those accumulator rows/cols are never stored).
__device__ __forceinline__ void load_tile_full(const bf16_t* __restrict__ G,
                                               int rows_total, int K, int row0, int k0,
                                               bf16_t* __restrict__ lds, int tid) {
  int r    = tid >> 1;
  int cseg = (tid & 1) * 16;
  int grow = row0 + r;
  bf16_t* dst = lds + r * BK + cseg;
  if (grow < rows_total) {
    const bf16_t* src = G + (size_t)grow * K + k0 + cseg;
#ifdef USE_ASYNC_LDS
    AS1 async_b64_t* gs = (AS1 async_b64_t*)(uintptr_t)src;
    AS3 async_b64_t* ls = (AS3 async_b64_t*)(uint32_t)(uintptr_t)dst;
    __builtin_amdgcn_global_load_async_to_lds_b64(gs, ls, 0, 0);
    __builtin_amdgcn_global_load_async_to_lds_b64(gs, ls, 8, 0);
    __builtin_amdgcn_global_load_async_to_lds_b64(gs, ls, 16, 0);
    __builtin_amdgcn_global_load_async_to_lds_b64(gs, ls, 24, 0);
#else
#pragma unroll
    for (int j = 0; j < 4; ++j)
      *(uint2*)(dst + j * 4) = *(const uint2*)(src + j * 4);
#endif
  }
}

// Partial K tile: zero-filled synchronous copy (correctness for K % 32 != 0).
__device__ __forceinline__ void load_tile_guard(const bf16_t* __restrict__ G,
                                                int rows_total, int K, int row0, int k0,
                                                bf16_t* __restrict__ lds, int tid) {
  int r    = tid >> 1;
  int cseg = (tid & 1) * 16;
  int grow = row0 + r;
  bf16_t* dst = lds + r * BK + cseg;
  if (grow < rows_total) {
    const bf16_t* src = G + (size_t)grow * K + k0 + cseg;
#pragma unroll
    for (int j = 0; j < 4; ++j) {
      int k = k0 + cseg + j * 4;
      if (k + 3 < K) {
        *(uint2*)(dst + j * 4) = *(const uint2*)(src + j * 4);
      } else {
#pragma unroll
        for (int e = 0; e < 4; ++e) {
          bf16_t v = (bf16_t)0.0f;
          if (k + e < K) v = src[j * 4 + e];
          dst[j * 4 + e] = v;
        }
      }
    }
  } else {
#pragma unroll
    for (int j = 0; j < 16; j += 4) *(uint2*)(dst + j) = make_uint2(0u, 0u);
  }
}

__device__ __forceinline__ bfx16 frag_ld(const bf16_t* __restrict__ base, int rbase,
                                         int lane) {
  // 16-bit A/B WMMA layout: lanes 0-15 -> K 0..7 & 16..23, lanes 16-31 -> K 8..15 & 24..31
  int lrow = lane & 15;
  int hs   = (lane >> 4) * 8;
  const bf16_t* p = base + (rbase + lrow) * BK + hs;
  union { bfx16 v; bfx8 h[2]; } u;
  u.h[0] = *(const bfx8*)p;
  u.h[1] = *(const bfx8*)(p + 16);
  return u.v;
}

__global__ __launch_bounds__(256, 1) void k_gemm_bf16_wmma(
    const bf16_t* __restrict__ A,   // M x K
    const bf16_t* __restrict__ Bt,  // N x K
    const float*  __restrict__ bias,// N
    float*        __restrict__ Cf,  // optional fp32 out (M x N)
    bf16_t*       __restrict__ Cb,  // optional bf16 out (M x N)
    int M, int N, int K, int relu) {
  __shared__ __align__(16) bf16_t As[2][BM * BK];
  __shared__ __align__(16) bf16_t Bs[2][BN * BK];

  int tid  = threadIdx.x;
  int lane = tid & 31;
  int wave = tid >> 5;
  int wm   = wave >> 2;   // 0..1 -> 64 rows each
  int wn   = wave & 3;    // 0..3 -> 32 cols each
  int m0   = blockIdx.y * BM;
  int n0   = blockIdx.x * BN;

  v8f zero = {0.f, 0.f, 0.f, 0.f, 0.f, 0.f, 0.f, 0.f};
  v8f acc[4][2];
#pragma unroll
  for (int i = 0; i < 4; ++i)
#pragma unroll
    for (int j = 0; j < 2; ++j) acc[i][j] = zero;

  int ktiles = (K + BK - 1) / BK;

  auto stage = [&](int kt, bf16_t* as, bf16_t* bs) {
    int k0 = kt * BK;
    if (k0 + BK <= K) {
      load_tile_full(A, M, K, m0, k0, as, tid);
      load_tile_full(Bt, N, K, n0, k0, bs, tid);
    } else {
      load_tile_guard(A, M, K, m0, k0, as, tid);
      load_tile_guard(Bt, N, K, n0, k0, bs, tid);
    }
  };

  stage(0, As[0], Bs[0]);
#ifdef USE_ASYNC_LDS
  WAIT_ASYNC();
#endif
  __syncthreads();

  int buf = 0;
  for (int kt = 0; kt < ktiles; ++kt) {
    if (kt + 1 < ktiles) stage(kt + 1, As[buf ^ 1], Bs[buf ^ 1]);

    bfx16 af[4], bfr[2];
#pragma unroll
    for (int i = 0; i < 4; ++i) af[i] = frag_ld(As[buf], wm * 64 + i * 16, lane);
#pragma unroll
    for (int j = 0; j < 2; ++j) bfr[j] = frag_ld(Bs[buf], wn * 32 + j * 16, lane);
#pragma unroll
    for (int i = 0; i < 4; ++i)
#pragma unroll
      for (int j = 0; j < 2; ++j)
        acc[i][j] = __builtin_amdgcn_wmma_f32_16x16x32_bf16(
            false, af[i], false, bfr[j], (short)0, acc[i][j], false, false);

#ifdef USE_ASYNC_LDS
    WAIT_ASYNC();          // my async LDS writes done before publishing buffer
#endif
    __syncthreads();
    buf ^= 1;
  }

  // Epilogue: C/D layout — VGPR r: lanes 0-15 -> M=r, lanes 16-31 -> M=r+8; N = lane&15
  int lcol  = lane & 15;
  int lrsel = (lane >> 4) * 8;
#pragma unroll
  for (int i = 0; i < 4; ++i) {
#pragma unroll
    for (int j = 0; j < 2; ++j) {
      int col = n0 + wn * 32 + j * 16 + lcol;
      if (col < N) {
        float bia = bias ? bias[col] : 0.f;
#pragma unroll
        for (int r = 0; r < 8; ++r) {
          int row = m0 + wm * 64 + i * 16 + lrsel + r;
          if (row < M) {
            float v = acc[i][j][r] + bia;
            if (relu) v = fmaxf(v, 0.f);
            if (Cf) Cf[(size_t)row * N + col] = v;
            if (Cb) Cb[(size_t)row * N + col] = (bf16_t)v;
          }
        }
      }
    }
  }
}

// ---------------------------------------------------------------------------
// BatchNorm (training-mode, biased var). One block per channel.
// If we != nullptr: out = bn * we[:,view], and vsum += out (fused).
// ---------------------------------------------------------------------------
__global__ __launch_bounds__(256) void k_bn(const float* __restrict__ Z,
                                            const float* __restrict__ g,
                                            const float* __restrict__ beta,
                                            float* __restrict__ out,
                                            float* __restrict__ vsum,
                                            const float* __restrict__ we,
                                            int view, int B_, int C) {
  int c = blockIdx.x;
  int t = threadIdx.x;
  float s = 0.f, s2 = 0.f;
  for (int r = t; r < B_; r += 256) {
    float v = Z[(size_t)r * C + c];
    s += v; s2 += v * v;
  }
  __shared__ float rs[256], rq[256];
  rs[t] = s; rq[t] = s2;
  __syncthreads();
  for (int off = 128; off > 0; off >>= 1) {
    if (t < off) { rs[t] += rs[t + off]; rq[t] += rq[t + off]; }
    __syncthreads();
  }
  float mean = rs[0] / (float)B_;
  float var  = rq[0] / (float)B_ - mean * mean;
  float inv  = rsqrtf(var + 1e-5f);
  float gg = g[c], bb = beta[c];
  for (int r = t; r < B_; r += 256) {
    float v = (Z[(size_t)r * C + c] - mean) * inv * gg + bb;
    if (we) {
      v *= we[(size_t)r * 3 + view];
      vsum[(size_t)r * C + c] += v;
    }
    out[(size_t)r * C + c] = v;
  }
}

// ---------------------------------------------------------------------------
// Head: p_v = sigmoid(relu(z_v) @ reg1W + b), conf, cm1, s_z. One thread/row.
// ---------------------------------------------------------------------------
__device__ __forceinline__ float view_head(const float* __restrict__ z,
                                           const float* __restrict__ Ws,
                                           const float* __restrict__ bs,
                                           float* __restrict__ pout) {
  float acc[20];
#pragma unroll
  for (int l = 0; l < 20; ++l) acc[l] = bs[l];
  for (int k = 0; k < 128; ++k) {
    float zv = fmaxf(z[k], 0.f);
#pragma unroll
    for (int l = 0; l < 20; ++l) acc[l] += zv * Ws[k * 20 + l];
  }
  const float denom = 1.f - __expf(-3.f);
  float csum = 0.f;
#pragma unroll
  for (int l = 0; l < 20; ++l) {
    float p = 1.f / (1.f + __expf(-acc[l]));
    pout[l] = p;
    csum += (1.f - __expf(-3.f * fabsf(2.f * p - 1.f))) / denom;
  }
  return csum * (1.f / 20.f);
}

__global__ __launch_bounds__(256) void k_head(
    const float* __restrict__ z0, const float* __restrict__ z1,
    const float* __restrict__ z2, const float* __restrict__ W,
    const float* __restrict__ bvec, const float* __restrict__ we,
    float* __restrict__ p0, float* __restrict__ p1, float* __restrict__ p2,
    float* __restrict__ s_z, int B_) {
  __shared__ float Ws[128 * 20];
  __shared__ float bs[20];
  for (int i = threadIdx.x; i < 128 * 20; i += 256) Ws[i] = W[i];
  if (threadIdx.x < 20) bs[threadIdx.x] = bvec[threadIdx.x];
  __syncthreads();
  int b = blockIdx.x * 256 + threadIdx.x;
  if (b >= B_) return;
  const float* za = z0 + (size_t)b * 128;
  const float* zb = z1 + (size_t)b * 128;
  const float* zc = z2 + (size_t)b * 128;
  float conf0 = view_head(za, Ws, bs, p0 + (size_t)b * 20);
  float conf1 = view_head(zb, Ws, bs, p1 + (size_t)b * 20);
  float conf2 = view_head(zc, Ws, bs, p2 + (size_t)b * 20);
  float w0 = we[(size_t)b * 3 + 0], w1 = we[(size_t)b * 3 + 1], w2 = we[(size_t)b * 3 + 2];
  float cm0 = conf0 * w0, cm1v = conf1 * w1, cm2 = conf2 * w2;
  float tot = cm0 + cm1v + cm2;
  float a0 = cm0 / tot * w0, a1 = cm1v / tot * w1, a2 = cm2 / tot * w2;
  float* sz = s_z + (size_t)b * 128;
  for (int k = 0; k < 128; ++k) sz[k] = a0 * za[k] + a1 * zb[k] + a2 * zc[k];
}

// ---------------------------------------------------------------------------
// Classifier MLP (128->20->20->4, tanh) over 4 inputs, scaled by result_matrix.
// ---------------------------------------------------------------------------
__device__ __forceinline__ void mlp_row(const float* __restrict__ in,
                                        const float* __restrict__ W0, const float* __restrict__ b0,
                                        const float* __restrict__ W1, const float* __restrict__ b1,
                                        const float* __restrict__ W2, const float* __restrict__ b2,
                                        float scale, float* __restrict__ out) {
  float h1[20];
#pragma unroll
  for (int l = 0; l < 20; ++l) h1[l] = b0[l];
  for (int k = 0; k < 128; ++k) {
    float v = in[k];
#pragma unroll
    for (int l = 0; l < 20; ++l) h1[l] += v * W0[k * 20 + l];
  }
#pragma unroll
  for (int l = 0; l < 20; ++l) h1[l] = tanhf(h1[l]);
  float h2[20];
#pragma unroll
  for (int l = 0; l < 20; ++l) h2[l] = b1[l];
#pragma unroll
  for (int k = 0; k < 20; ++k)
#pragma unroll
    for (int l = 0; l < 20; ++l) h2[l] += h1[k] * W1[k * 20 + l];
#pragma unroll
  for (int l = 0; l < 20; ++l) h2[l] = tanhf(h2[l]);
  float o[4];
#pragma unroll
  for (int c = 0; c < 4; ++c) o[c] = b2[c];
#pragma unroll
  for (int k = 0; k < 20; ++k)
#pragma unroll
    for (int c = 0; c < 4; ++c) o[c] += h2[k] * W2[k * 4 + c];
#pragma unroll
  for (int c = 0; c < 4; ++c) out[c] = tanhf(o[c]) * scale;
}

__global__ __launch_bounds__(256) void k_vres(
    const float* __restrict__ vz0, const float* __restrict__ vz1,
    const float* __restrict__ vz2, const float* __restrict__ s_z,
    const float* __restrict__ W0, const float* __restrict__ b0,
    const float* __restrict__ W1, const float* __restrict__ b1,
    const float* __restrict__ W2, const float* __restrict__ b2,
    const float* __restrict__ we,
    float* __restrict__ r0, float* __restrict__ r1,
    float* __restrict__ r2, float* __restrict__ r3, int B_) {
  __shared__ float sW0[128 * 20], sW1[20 * 20], sW2[20 * 4];
  __shared__ float sb0[20], sb1[20], sb2[4];
  for (int i = threadIdx.x; i < 128 * 20; i += 256) sW0[i] = W0[i];
  for (int i = threadIdx.x; i < 20 * 20; i += 256)  sW1[i] = W1[i];
  for (int i = threadIdx.x; i < 20 * 4; i += 256)   sW2[i] = W2[i];
  if (threadIdx.x < 20) { sb0[threadIdx.x] = b0[threadIdx.x]; sb1[threadIdx.x] = b1[threadIdx.x]; }
  if (threadIdx.x < 4)  sb2[threadIdx.x] = b2[threadIdx.x];
  __syncthreads();
  int b = blockIdx.x * 256 + threadIdx.x;
  if (b >= B_) return;
  float w0 = we[(size_t)b * 3 + 0], w1 = we[(size_t)b * 3 + 1], w2 = we[(size_t)b * 3 + 2];
  mlp_row(vz0 + (size_t)b * 128, sW0, sb0, sW1, sb1, sW2, sb2, w0, r0 + (size_t)b * 4);
  mlp_row(vz1 + (size_t)b * 128, sW0, sb0, sW1, sb1, sW2, sb2, w1, r1 + (size_t)b * 4);
  mlp_row(vz2 + (size_t)b * 128, sW0, sb0, sW1, sb1, sW2, sb2, w2, r2 + (size_t)b * 4);
  mlp_row(s_z + (size_t)b * 128, sW0, sb0, sW1, sb1, sW2, sb2, 1.f, r3 + (size_t)b * 4);
}

// ---------------------------------------------------------------------------
// Final gate: z = relu(s_z * sigmoid(v_z)); yLabel = sigmoid(z @ regW + regb)
// ---------------------------------------------------------------------------
__global__ __launch_bounds__(256) void k_final(
    const float* __restrict__ s_z, const float* __restrict__ v_z,
    const float* __restrict__ W, const float* __restrict__ bvec,
    float* __restrict__ z_out, float* __restrict__ y_out, int B_) {
  __shared__ float Ws[128 * 20];
  __shared__ float bs[20];
  for (int i = threadIdx.x; i < 128 * 20; i += 256) Ws[i] = W[i];
  if (threadIdx.x < 20) bs[threadIdx.x] = bvec[threadIdx.x];
  __syncthreads();
  int b = blockIdx.x * 256 + threadIdx.x;
  if (b >= B_) return;
  float acc[20];
#pragma unroll
  for (int l = 0; l < 20; ++l) acc[l] = bs[l];
  for (int k = 0; k < 128; ++k) {
    float sv = s_z[(size_t)b * 128 + k];
    float vv = v_z[(size_t)b * 128 + k];
    float zz = fmaxf(sv * (1.f / (1.f + __expf(-vv))), 0.f);
    z_out[(size_t)b * 128 + k] = zz;
#pragma unroll
    for (int l = 0; l < 20; ++l) acc[l] += zz * Ws[k * 20 + l];
  }
#pragma unroll
  for (int l = 0; l < 20; ++l)
    y_out[(size_t)b * 20 + l] = 1.f / (1.f + __expf(-acc[l]));
}

// ---------------------------------------------------------------------------
// Host orchestration
// ---------------------------------------------------------------------------
static inline size_t align256(size_t x) { return (x + 255) & ~(size_t)255; }

extern "C" void kernel_launch(void* const* d_in, const int* in_sizes, int n_in,
                              void* d_out, int out_size, void* d_ws, size_t ws_size,
                              hipStream_t stream) {
  (void)n_in; (void)out_size; (void)ws_size;
  const int B = 8192, NZ = 128, NL = 20;
  const int D1 = 1600, D2 = 1600, D3 = 1500;

  // ---- input map (jax tree_leaves order: dict keys sorted) -----------------
  // 0:x0 1:x1 2:x2 3:we
  // params sorted keys: cls, enc1, enc2, reg, reg1, vv_b, vv_g
  // cls: 4..9 (W0,b0,W1,b1,W2,b2)
  // enc dicts sorted keys: W1,W2,W3,Wz,b1,b2,b3,beta,bz,g
  // enc1[v]: 10+10v .. ; enc2[v]: 40+10v .. ; reg:70,71 ; reg1:72,73 ; vv_b:74 ; vv_g:75
  const float* we    = (const float*)d_in[3];
  const float* clsW0 = (const float*)d_in[4]; const float* clsb0 = (const float*)d_in[5];
  const float* clsW1 = (const float*)d_in[6]; const float* clsb1 = (const float*)d_in[7];
  const float* clsW2 = (const float*)d_in[8]; const float* clsb2 = (const float*)d_in[9];
  const float* regW  = (const float*)d_in[70]; const float* regb  = (const float*)d_in[71];
  const float* reg1W = (const float*)d_in[72]; const float* reg1b = (const float*)d_in[73];
  const float* vv_b  = (const float*)d_in[74]; const float* vv_g  = (const float*)d_in[75];

  // ---- workspace layout ----------------------------------------------------
  char* w = (char*)d_ws;
  bf16_t* xb  = (bf16_t*)w; w += align256((size_t)B * 2000 * sizeof(bf16_t));
  bf16_t* Wt  = (bf16_t*)w; w += align256((size_t)2000 * 1600 * sizeof(bf16_t));
  bf16_t* hA  = (bf16_t*)w; w += align256((size_t)B * 1600 * sizeof(bf16_t));
  bf16_t* hB  = (bf16_t*)w; w += align256((size_t)B * 1600 * sizeof(bf16_t));
  float* zbuf = (float*)w;  w += align256((size_t)B * NZ * sizeof(float));
  float* s_z  = (float*)w;  w += align256((size_t)B * NZ * sizeof(float));
  float* vsum = (float*)w;  w += align256((size_t)B * NZ * sizeof(float));
  float* v_z  = (float*)w;  w += align256((size_t)B * NZ * sizeof(float));

  // ---- output layout (flat, return order) ----------------------------------
  float* out = (float*)d_out;
  size_t off = 0;
  float* o_y = out + off;  off += (size_t)B * NL;
  float* o_z = out + off;  off += (size_t)B * NZ;
  float* o_sz[3]; for (int v = 0; v < 3; ++v) { o_sz[v] = out + off; off += (size_t)B * NZ; }
  float* o_vz[3]; for (int v = 0; v < 3; ++v) { o_vz[v] = out + off; off += (size_t)B * NZ; }
  float* o_p[3];  for (int v = 0; v < 3; ++v) { o_p[v]  = out + off; off += (size_t)B * NL; }
  float* o_vr[4]; for (int v = 0; v < 4; ++v) { o_vr[v] = out + off; off += (size_t)B * 4; }

  k_zero_f32<<<4096, 256, 0, stream>>>(vsum, B * NZ);

  auto cvtT = [&](const float* src, int K, int N) {
    dim3 g((N + 31) / 32, (K + 31) / 32);
    k_cvtT_bf16<<<g, 256, 0, stream>>>(src, Wt, K, N);
  };
  auto gemm = [&](const bf16_t* A, const float* bias, float* Cf, bf16_t* Cb,
                  int M, int N, int K, int relu) {
    dim3 g((N + BN - 1) / BN, (M + BM - 1) / BM);
    k_gemm_bf16_wmma<<<g, 256, 0, stream>>>(A, Wt, bias, Cf, Cb, M, N, K, relu);
  };
  auto run_encoder = [&](int n_dim, void* const* P) {
    const float* W1 = (const float*)P[0]; const float* W2 = (const float*)P[1];
    const float* W3 = (const float*)P[2]; const float* Wz = (const float*)P[3];
    const float* b1 = (const float*)P[4]; const float* b2 = (const float*)P[5];
    const float* b3 = (const float*)P[6]; const float* bz = (const float*)P[8];
    cvtT(W1, n_dim, D1); gemm(xb, b1, nullptr, hA, B, D1, n_dim, 1);
    cvtT(W2, D1, D2);    gemm(hA, b2, nullptr, hB, B, D2, D1, 1);
    cvtT(W3, D2, D3);    gemm(hB, b3, nullptr, hA, B, D3, D2, 1);
    cvtT(Wz, D3, NZ);    gemm(hA, bz, zbuf, nullptr, B, NZ, D3, 0);
  };

  for (int v = 0; v < 3; ++v) {
    int n_dim = in_sizes[v] / B;
    const float* xv = (const float*)d_in[v];
    k_cvt_bf16<<<8192, 256, 0, stream>>>(xv, xb, B * n_dim);

    // shared encoder -> batchnorm -> share_z output
    run_encoder(n_dim, &d_in[10 + 10 * v]);
    {
      const float* g  = (const float*)d_in[10 + 10 * v + 9];
      const float* be = (const float*)d_in[10 + 10 * v + 7];
      k_bn<<<NZ, 256, 0, stream>>>(zbuf, g, be, o_sz[v], nullptr, nullptr, 0, B, NZ);
    }
    // view-specific encoder -> batchnorm * we[:,v] -> viewsp output (+vsum)
    run_encoder(n_dim, &d_in[40 + 10 * v]);
    {
      const float* g  = (const float*)d_in[40 + 10 * v + 9];
      const float* be = (const float*)d_in[40 + 10 * v + 7];
      k_bn<<<NZ, 256, 0, stream>>>(zbuf, g, be, o_vz[v], vsum, we, v, B, NZ);
    }
  }

  // v_z = bn(sum viewsp, vv_g, vv_b)
  k_bn<<<NZ, 256, 0, stream>>>(vsum, vv_g, vv_b, v_z, nullptr, nullptr, 0, B, NZ);

  // p_list, conf fusion, s_z
  k_head<<<(B + 255) / 256, 256, 0, stream>>>(o_sz[0], o_sz[1], o_sz[2], reg1W, reg1b,
                                              we, o_p[0], o_p[1], o_p[2], s_z, B);
  // classifier heads
  k_vres<<<(B + 255) / 256, 256, 0, stream>>>(o_vz[0], o_vz[1], o_vz[2], s_z,
                                              clsW0, clsb0, clsW1, clsb1, clsW2, clsb2,
                                              we, o_vr[0], o_vr[1], o_vr[2], o_vr[3], B);
  // gate + label regressor
  k_final<<<(B + 255) / 256, 256, 0, stream>>>(s_z, v_z, regW, regb, o_z, o_y, B);
}
